// TransformerBlock_910533066930
// MI455X (gfx1250) — matrix operand order repaired and checked by reading, not compiled, not run on hardware
//
#include <hip/hip_runtime.h>
#include <hip/hip_bf16.h>

// ---------------------------------------------------------------------------
// Transformer block (pre-norm, RoPE, causal attention, SwiGLU) for gfx1250.
// GEMMs: v_wmma_f32_16x16x32_bf16 with weight tiles staged to LDS by the
// Tensor Data Mover (tensor_load_to_lds, double-buffered, TENSORcnt-synced).
// Attention: flash-style with WMMA scores + LDS-reshaped P + WMMA P*V.
// ---------------------------------------------------------------------------

typedef __bf16 bf16_t;
typedef __attribute__((ext_vector_type(16))) __bf16 v16bf;
typedef __attribute__((ext_vector_type(8)))  __bf16 v8bf;
typedef __attribute__((ext_vector_type(8)))  float  v8f;
typedef unsigned int u32x4 __attribute__((ext_vector_type(4)));
typedef int          i32x8 __attribute__((ext_vector_type(8)));
typedef int          i32x4 __attribute__((ext_vector_type(4)));

#define D_MODEL 1024
#define D_FF    2752
#define SEQ     2048
#define NTOK    4096      // B*S
#define NHEAD   16
#define DK      64

// LDS weight tile: 64 N-rows x 64 K-elements bf16, padded 8 elems/row by TDM
#define TILE_K      64
#define LDS_STRIDE  72    // 64 + 8 pad elements (144 B rows, 16B aligned)

__device__ __forceinline__ v8f zero_v8f() {
  v8f z;
#pragma unroll
  for (int i = 0; i < 8; ++i) z[i] = 0.f;
  return z;
}

__device__ __forceinline__ v8f wmma_bf16(v16bf a, v16bf b, v8f c) {
  return __builtin_amdgcn_wmma_f32_16x16x32_bf16(
      false, a, false, b, (short)0, c, false, false);
}

// A-matrix 16x32 bf16 fragment: lane row = lane%16;
// element e: K = e + 16*(lane/16)
__device__ __forceinline__ v16bf load_a_frag(const bf16_t* base, int ldk, int lane) {
  const bf16_t* p = base + (size_t)(lane & 15) * ldk + ((lane & 16) ? 16 : 0);
  v8bf lo = *(const v8bf*)p;
  v8bf hi = *(const v8bf*)(p + 8);
  v16bf r;
#pragma unroll
  for (int i = 0; i < 8; ++i) { r[i] = lo[i]; r[i + 8] = hi[i]; }
  return r;
}

// B-matrix 32x16 fragment from global: lane col n = lane%16, K = e + 16*(lane/16)
__device__ __forceinline__ v16bf load_b_frag(const bf16_t* base, int ldk, int lane) {
  const bf16_t* p = base + (size_t)(lane & 15) * ldk + ((lane & 16) ? 16 : 0);
  return *(const v16bf*)p;
}

// B-matrix 32x16 fragment from a TDM-staged LDS tile (row stride LDS_STRIDE)
__device__ __forceinline__ v16bf load_b_frag_lds(const bf16_t* tile, int row0,
                                                 int kk, int lane) {
  const bf16_t* p = tile + (row0 + (lane & 15)) * LDS_STRIDE + kk +
                    ((lane & 16) ? 16 : 0);
  v8bf lo = *(const v8bf*)p;
  v8bf hi = *(const v8bf*)(p + 8);
  v16bf r;
#pragma unroll
  for (int i = 0; i < 8; ++i) { r[i] = lo[i]; r[i + 8] = hi[i]; }
  return r;
}

// ---------------------------------------------------------------------------
// Tensor Data Mover: DMA one 64(N) x 64(K) bf16 tile of W[N,K] into LDS.
// D# per CDNA5 ISA. data_size=2B; pad_enable with
// pad_interval=4 (128B = one row) and pad_amount=3 (16B) -> LDS stride 72 elems.
// ---------------------------------------------------------------------------
__device__ __forceinline__ void tdm_load_tile(const bf16_t* gsrc,
                                              unsigned lds_addr, int K, int N) {
  unsigned long long ga = (unsigned long long)(uintptr_t)gsrc;
  u32x4 g0;
  g0[0] = 1u;                                  // count=1, user mode
  g0[1] = lds_addr;                            // lds_addr [63:32]
  g0[2] = (unsigned)(ga & 0xFFFFFFFFu);        // global_addr [95:64]
  g0[3] = (unsigned)(ga >> 32) | (2u << 30);   // addr[56:32] | type=2
  i32x8 g1;
  g1[0] = (1 << 16) | (1 << 20) | (4 << 22) | (3 << 25);
  //       data_size=2B | pad_en | pad_interval=32DW | pad_amount=4DW
  g1[1] = (K & 0xFFFF) << 16;                  // tensor_dim0 lo (atomic_bar=0)
  g1[2] = ((N & 0xFFFF) << 16) | ((K >> 16) & 0xFFFF);   // dim1 lo | dim0 hi
  g1[3] = (TILE_K << 16) | ((N >> 16) & 0xFFFF);         // tile_dim0 | dim1 hi
  g1[4] = 64;                                  // tile_dim1=64, tile_dim2=0
  g1[5] = K;                                   // tensor_dim0_stride lo
  g1[6] = 0;                                   // stride0 hi | dim1_stride lo
  g1[7] = 0;
  i32x4 z4 = {0, 0, 0, 0};
#if defined(__clang_major__) && (__clang_major__ >= 23)
  i32x8 z8 = {0, 0, 0, 0, 0, 0, 0, 0};
  __builtin_amdgcn_tensor_load_to_lds(g0, g1, z4, z4, z8, 0);
#else
  __builtin_amdgcn_tensor_load_to_lds(g0, g1, z4, z4, 0);
#endif
}

__device__ __forceinline__ unsigned lds_off32(const void* p) {
  return (unsigned)(uintptr_t)p;   // aperture: LDS offset = addr[31:0]
}

// ---------------------------------------------------------------------------
// fp32 -> bf16 bulk convert (weights)
// ---------------------------------------------------------------------------
__global__ __launch_bounds__(256) void cvt_bf16_kernel(
    const float* __restrict__ s, bf16_t* __restrict__ d, int n4) {
  int i = blockIdx.x * 256 + threadIdx.x;
  if (i < n4) {
    float4 v = ((const float4*)s)[i];
    bf16_t* p = d + (size_t)i * 4;
    p[0] = (bf16_t)v.x; p[1] = (bf16_t)v.y;
    p[2] = (bf16_t)v.z; p[3] = (bf16_t)v.w;
  }
}

// ---------------------------------------------------------------------------
// RMSNorm: one block per token row (1024 cols), fp32 in -> bf16 out
// ---------------------------------------------------------------------------
__global__ __launch_bounds__(256) void rmsnorm_kernel(
    const float* __restrict__ x, const float* __restrict__ w,
    bf16_t* __restrict__ out) {
  const int row = blockIdx.x;
  const float4 v = ((const float4*)(x + (size_t)row * D_MODEL))[threadIdx.x];
  float ss = v.x * v.x + v.y * v.y + v.z * v.z + v.w * v.w;
#pragma unroll
  for (int off = 16; off >= 1; off >>= 1) ss += __shfl_xor(ss, off, 32);
  __shared__ float red[8];
  if ((threadIdx.x & 31) == 0) red[threadIdx.x >> 5] = ss;
  __syncthreads();
  float tot = 0.f;
#pragma unroll
  for (int i = 0; i < 8; ++i) tot += red[i];
  const float inv = rsqrtf(tot * (1.0f / D_MODEL) + 1e-5f);
  const float4 wv = ((const float4*)w)[threadIdx.x];
  bf16_t* op = out + (size_t)row * D_MODEL + threadIdx.x * 4;
  op[0] = (bf16_t)(v.x * inv * wv.x);
  op[1] = (bf16_t)(v.y * inv * wv.y);
  op[2] = (bf16_t)(v.z * inv * wv.z);
  op[3] = (bf16_t)(v.w * inv * wv.w);
}

// ---------------------------------------------------------------------------
// Generic GEMM: C(f32)[M,N] = A(bf16)[M,K] * W(bf16)[N,K]^T + resid(f32)
// Block: 8 waves x (32M x 64N) = 256M x 64N. W tiles TDM-staged, dbl-buffered.
// ---------------------------------------------------------------------------
__global__ __launch_bounds__(256) void gemm_rc_f32_kernel(
    const bf16_t* __restrict__ A, const bf16_t* __restrict__ W,
    const float* __restrict__ resid, float* __restrict__ C,
    int N, int K) {
  __shared__ __attribute__((aligned(256))) bf16_t btile[2][64 * LDS_STRIDE];
  const int lane = threadIdx.x & 31;
  const int wave = threadIdx.x >> 5;
  const int n0 = blockIdx.x * 64;
  const int m0 = blockIdx.y * 256 + wave * 32;

  v8f acc[2][4];
#pragma unroll
  for (int mi = 0; mi < 2; ++mi)
#pragma unroll
    for (int t = 0; t < 4; ++t) acc[mi][t] = zero_v8f();

  const bf16_t* ab0 = A + (size_t)m0 * K;
  const bf16_t* ab1 = A + (size_t)(m0 + 16) * K;
  const unsigned l0 = lds_off32(&btile[0][0]);
  const unsigned l1 = lds_off32(&btile[1][0]);
  const int nkb = K / TILE_K;

  if (wave == 0) {
    tdm_load_tile(W + (size_t)n0 * K, l0, K, N);
    __builtin_amdgcn_s_wait_tensorcnt(0);
  }
  __syncthreads();

  int buf = 0;
  for (int kb = 0; kb < nkb; ++kb) {
    if (wave == 0 && (kb + 1) < nkb)
      tdm_load_tile(W + (size_t)n0 * K + (size_t)(kb + 1) * TILE_K,
                    buf ? l0 : l1, K, N);
    const bf16_t* tl = &btile[buf][0];
#pragma unroll
    for (int s2 = 0; s2 < 2; ++s2) {
      const int k0 = kb * TILE_K + s2 * 32;
      __builtin_prefetch(ab0 + k0 + 512, 0, 1);
      v16bf a0 = load_a_frag(ab0 + k0, K, lane);
      v16bf a1 = load_a_frag(ab1 + k0, K, lane);
#pragma unroll
      for (int t = 0; t < 4; ++t) {
        v16bf bfr = load_b_frag_lds(tl, t * 16, s2 * 32, lane);
        acc[0][t] = wmma_bf16(a0, bfr, acc[0][t]);
        acc[1][t] = wmma_bf16(a1, bfr, acc[1][t]);
      }
    }
    if (wave == 0) __builtin_amdgcn_s_wait_tensorcnt(0);
    __syncthreads();
    buf ^= 1;
  }

  const int col = lane & 15;
  const int rb = (lane & 16) ? 8 : 0;
#pragma unroll
  for (int mi = 0; mi < 2; ++mi)
#pragma unroll
    for (int t = 0; t < 4; ++t)
#pragma unroll
      for (int r = 0; r < 8; ++r) {
        size_t idx = (size_t)(m0 + mi * 16 + rb + r) * N + (n0 + t * 16 + col);
        C[idx] = acc[mi][t][r] + resid[idx];
      }
}

// ---------------------------------------------------------------------------
// QKV projection GEMM (N=K=1024), TDM-staged weights, fused epilogue:
//   mode 0/1: RoPE -> bf16 [NTOK,1024];  mode 2: V transposed bf16 [B,H,64,S]
// ---------------------------------------------------------------------------
__global__ __launch_bounds__(256) void gemm_qkv_kernel(
    const bf16_t* __restrict__ A, const bf16_t* __restrict__ W,
    const int* __restrict__ tpos, bf16_t* __restrict__ out, int mode) {
  __shared__ __attribute__((aligned(256))) bf16_t btile[2][64 * LDS_STRIDE];
  const int K = D_MODEL;
  const int lane = threadIdx.x & 31;
  const int wave = threadIdx.x >> 5;
  const int n0 = blockIdx.x * 64;
  const int m0 = blockIdx.y * 256 + wave * 32;

  v8f acc[2][4];
#pragma unroll
  for (int mi = 0; mi < 2; ++mi)
#pragma unroll
    for (int t = 0; t < 4; ++t) acc[mi][t] = zero_v8f();

  const bf16_t* ab0 = A + (size_t)m0 * K;
  const bf16_t* ab1 = A + (size_t)(m0 + 16) * K;
  const unsigned l0 = lds_off32(&btile[0][0]);
  const unsigned l1 = lds_off32(&btile[1][0]);
  const int nkb = K / TILE_K;

  if (wave == 0) {
    tdm_load_tile(W + (size_t)n0 * K, l0, K, D_MODEL);
    __builtin_amdgcn_s_wait_tensorcnt(0);
  }
  __syncthreads();

  int buf = 0;
  for (int kb = 0; kb < nkb; ++kb) {
    if (wave == 0 && (kb + 1) < nkb)
      tdm_load_tile(W + (size_t)n0 * K + (size_t)(kb + 1) * TILE_K,
                    buf ? l0 : l1, K, D_MODEL);
    const bf16_t* tl = &btile[buf][0];
#pragma unroll
    for (int s2 = 0; s2 < 2; ++s2) {
      const int k0 = kb * TILE_K + s2 * 32;
      v16bf a0 = load_a_frag(ab0 + k0, K, lane);
      v16bf a1 = load_a_frag(ab1 + k0, K, lane);
#pragma unroll
      for (int t = 0; t < 4; ++t) {
        v16bf bfr = load_b_frag_lds(tl, t * 16, s2 * 32, lane);
        acc[0][t] = wmma_bf16(a0, bfr, acc[0][t]);
        acc[1][t] = wmma_bf16(a1, bfr, acc[1][t]);
      }
    }
    if (wave == 0) __builtin_amdgcn_s_wait_tensorcnt(0);
    __syncthreads();
    buf ^= 1;
  }

  const int col = lane & 15;
  const int rb = (lane & 16) ? 8 : 0;
  const float kLog = -9.210340371976184f / 32.0f;  // -ln(10000)/(DK/2)

#pragma unroll
  for (int mi = 0; mi < 2; ++mi)
#pragma unroll
    for (int t = 0; t < 4; ++t)
#pragma unroll
      for (int r = 0; r < 8; ++r) {
        const int grow = m0 + mi * 16 + rb + r;
        const int gc = n0 + t * 16 + col;
        float val = acc[mi][t][r];
        float partner = __shfl_xor(val, 1, 32);
        if (mode < 2) {
          const int d = gc & (DK - 1);
          const int pair = d >> 1;
          const float pos = (float)tpos[grow & (SEQ - 1)];
          const float ang = pos * __expf((float)pair * kLog);
          const float c = __cosf(ang), s = __sinf(ang);
          const float res = (d & 1) ? (val * c + partner * s)
                                    : (val * c - partner * s);
          out[(size_t)grow * D_MODEL + gc] = (bf16_t)res;
        } else {
          const int b = grow >> 11, sidx = grow & (SEQ - 1);
          const int h = gc >> 6, dd = gc & (DK - 1);
          out[(size_t)((b * NHEAD + h) * DK + dd) * SEQ + sidx] = (bf16_t)val;
        }
      }
}

// ---------------------------------------------------------------------------
// Flash attention: one wave per 16-query tile per (b, h). KV step = 32.
// ---------------------------------------------------------------------------
__global__ __launch_bounds__(256) void attn_kernel(
    const bf16_t* __restrict__ qbf, const bf16_t* __restrict__ kbf,
    const bf16_t* __restrict__ vt, bf16_t* __restrict__ obf) {
  __shared__ __attribute__((aligned(32))) bf16_t pbuf[8][16 * 40];

  const int lane = threadIdx.x & 31;
  const int wave = threadIdx.x >> 5;
  const int task = blockIdx.x * 8 + wave;
  const int qt = task & 127;
  const int h = (task >> 7) & (NHEAD - 1);
  const int b = task >> 11;
  const int q0 = qt * 16;
  const int col = lane & 15;
  const int rb = (lane & 16) ? 8 : 0;

  const bf16_t* qbase = qbf + (size_t)(b * SEQ + q0) * D_MODEL + h * DK;
  v16bf qf0 = load_a_frag(qbase, D_MODEL, lane);
  v16bf qf1 = load_a_frag(qbase + 32, D_MODEL, lane);

  v8f o[4];
#pragma unroll
  for (int t = 0; t < 4; ++t) o[t] = zero_v8f();
  float mstate[8], lstate[8];
#pragma unroll
  for (int r = 0; r < 8; ++r) { mstate[r] = -3.0e38f; lstate[r] = 0.f; }

  bf16_t* lp = &pbuf[wave][0];

  for (int j0 = 0; j0 <= q0 + 15; j0 += 32) {
    const bf16_t* kb0 = kbf + (size_t)(b * SEQ + j0) * D_MODEL + h * DK;
    const bf16_t* kb1 = kb0 + (size_t)16 * D_MODEL;

    v8f s0 = zero_v8f(), s1 = zero_v8f();
    s0 = wmma_bf16(qf0, load_b_frag(kb0, D_MODEL, lane), s0);
    s0 = wmma_bf16(qf1, load_b_frag(kb0 + 32, D_MODEL, lane), s0);
    s1 = wmma_bf16(qf0, load_b_frag(kb1, D_MODEL, lane), s1);
    s1 = wmma_bf16(qf1, load_b_frag(kb1 + 32, D_MODEL, lane), s1);

#pragma unroll
    for (int r = 0; r < 8; ++r) {
      const int qrow = q0 + rb + r;
      float a0 = s0[r] * 0.125f;
      float a1 = s1[r] * 0.125f;
      if (j0 + col > qrow)      a0 = -1.0e30f;
      if (j0 + 16 + col > qrow) a1 = -1.0e30f;
      float mx = fmaxf(a0, a1);
#pragma unroll
      for (int off = 8; off >= 1; off >>= 1) mx = fmaxf(mx, __shfl_xor(mx, off, 16));
      const float mnew = fmaxf(mstate[r], mx);
      const float corr = __expf(mstate[r] - mnew);
      const float p0 = __expf(a0 - mnew);
      const float p1 = __expf(a1 - mnew);
      float sum = p0 + p1;
#pragma unroll
      for (int off = 8; off >= 1; off >>= 1) sum += __shfl_xor(sum, off, 16);
      lstate[r] = lstate[r] * corr + sum;
      mstate[r] = mnew;
#pragma unroll
      for (int t = 0; t < 4; ++t) o[t][r] *= corr;
      lp[(rb + r) * 40 + col] = (bf16_t)p0;
      lp[(rb + r) * 40 + 16 + col] = (bf16_t)p1;
    }

    asm volatile("s_wait_dscnt 0" ::: "memory");

    v16bf pf = load_a_frag(lp, 40, lane);
    const bf16_t* vbase = vt + (size_t)((b * NHEAD + h) * DK) * SEQ + j0;
#pragma unroll
    for (int t = 0; t < 4; ++t)
      o[t] = wmma_bf16(pf, load_b_frag(vbase + (size_t)(t * 16) * SEQ, SEQ, lane), o[t]);
  }

#pragma unroll
  for (int t = 0; t < 4; ++t)
#pragma unroll
    for (int r = 0; r < 8; ++r) {
      const float val = o[t][r] / lstate[r];
      obf[(size_t)(b * SEQ + q0 + rb + r) * D_MODEL + h * DK + t * 16 + col] =
          (bf16_t)val;
    }
}

// ---------------------------------------------------------------------------
// Fused SwiGLU GEMM: F(bf16)[M,D_FF] = silu(A*W1^T) * (A*W3^T); K=1024.
// Two TDM streams (wave0: W1, wave1: W3), shared A-fragments.
// ---------------------------------------------------------------------------
__global__ __launch_bounds__(256) void gemm_ffn_gate_kernel(
    const bf16_t* __restrict__ A, const bf16_t* __restrict__ W1,
    const bf16_t* __restrict__ W3, bf16_t* __restrict__ F) {
  __shared__ __attribute__((aligned(256))) bf16_t t1[2][64 * LDS_STRIDE];
  __shared__ __attribute__((aligned(256))) bf16_t t3[2][64 * LDS_STRIDE];
  const int K = D_MODEL;
  const int lane = threadIdx.x & 31;
  const int wave = threadIdx.x >> 5;
  const int n0 = blockIdx.x * 64;
  const int m0 = blockIdx.y * 128 + wave * 16;

  v8f accu[4], accg[4];
#pragma unroll
  for (int t = 0; t < 4; ++t) { accu[t] = zero_v8f(); accg[t] = zero_v8f(); }

  const bf16_t* abase = A + (size_t)m0 * K;
  const unsigned l1a = lds_off32(&t1[0][0]);
  const unsigned l1b = lds_off32(&t1[1][0]);
  const unsigned l3a = lds_off32(&t3[0][0]);
  const unsigned l3b = lds_off32(&t3[1][0]);
  const int nkb = K / TILE_K;

  if (wave == 0) tdm_load_tile(W1 + (size_t)n0 * K, l1a, K, D_FF);
  if (wave == 1) tdm_load_tile(W3 + (size_t)n0 * K, l3a, K, D_FF);
  if (wave < 2) __builtin_amdgcn_s_wait_tensorcnt(0);
  __syncthreads();

  int buf = 0;
  for (int kb = 0; kb < nkb; ++kb) {
    if ((kb + 1) < nkb) {
      const size_t woff = (size_t)n0 * K + (size_t)(kb + 1) * TILE_K;
      if (wave == 0) tdm_load_tile(W1 + woff, buf ? l1a : l1b, K, D_FF);
      if (wave == 1) tdm_load_tile(W3 + woff, buf ? l3a : l3b, K, D_FF);
    }
    const bf16_t* tu = &t1[buf][0];
    const bf16_t* tg = &t3[buf][0];
#pragma unroll
    for (int s2 = 0; s2 < 2; ++s2) {
      const int k0 = kb * TILE_K + s2 * 32;
      v16bf af = load_a_frag(abase + k0, K, lane);
#pragma unroll
      for (int t = 0; t < 4; ++t) {
        accu[t] = wmma_bf16(af, load_b_frag_lds(tu, t * 16, s2 * 32, lane), accu[t]);
        accg[t] = wmma_bf16(af, load_b_frag_lds(tg, t * 16, s2 * 32, lane), accg[t]);
      }
    }
    if (wave < 2) __builtin_amdgcn_s_wait_tensorcnt(0);
    __syncthreads();
    buf ^= 1;
  }

  const int col = lane & 15;
  const int rb = (lane & 16) ? 8 : 0;
#pragma unroll
  for (int t = 0; t < 4; ++t)
#pragma unroll
    for (int r = 0; r < 8; ++r) {
      const float u = accu[t][r];
      const float g = accg[t][r];
      const float f = (u / (1.f + __expf(-u))) * g;
      F[(size_t)(m0 + rb + r) * D_FF + (n0 + t * 16 + col)] = (bf16_t)f;
    }
}

// ---------------------------------------------------------------------------
extern "C" void kernel_launch(void* const* d_in, const int* in_sizes, int n_in,
                              void* d_out, int out_size, void* d_ws, size_t ws_size,
                              hipStream_t stream) {
  (void)in_sizes; (void)n_in; (void)out_size; (void)ws_size;
  const float* x  = (const float*)d_in[0];
  const int* tpos = (const int*)d_in[1];
  const float* wq = (const float*)d_in[2];
  const float* wk = (const float*)d_in[3];
  const float* wv = (const float*)d_in[4];
  const float* wo = (const float*)d_in[5];
  const float* n1 = (const float*)d_in[6];
  const float* n2 = (const float*)d_in[7];
  const float* w1 = (const float*)d_in[8];
  const float* w2 = (const float*)d_in[9];
  const float* w3 = (const float*)d_in[10];

  char* ws = (char*)d_ws;
  size_t off = 0;
  auto abf = [&](size_t elems) {
    bf16_t* p = (bf16_t*)(ws + off);
    off += elems * sizeof(bf16_t);
    return p;
  };
  const size_t SQ = (size_t)D_MODEL * D_MODEL;
  const size_t SF = (size_t)D_FF * D_MODEL;
  const size_t SA = (size_t)NTOK * D_MODEL;
  bf16_t* wq_b = abf(SQ); bf16_t* wk_b = abf(SQ);
  bf16_t* wv_b = abf(SQ); bf16_t* wo_b = abf(SQ);
  bf16_t* w1_b = abf(SF); bf16_t* w3_b = abf(SF); bf16_t* w2_b = abf(SF);
  bf16_t* h_b  = abf(SA); bf16_t* q_b  = abf(SA); bf16_t* k_b = abf(SA);
  bf16_t* vt_b = abf(SA); bf16_t* at_b = abf(SA); bf16_t* h2_b = abf(SA);
  bf16_t* f_b  = abf((size_t)NTOK * D_FF);
  float* y = (float*)(ws + off);   // fp32 residual stream (~115 MB total)

  cvt_bf16_kernel<<<(int)(SQ / 4 + 255) / 256, 256, 0, stream>>>(wq, wq_b, (int)(SQ / 4));
  cvt_bf16_kernel<<<(int)(SQ / 4 + 255) / 256, 256, 0, stream>>>(wk, wk_b, (int)(SQ / 4));
  cvt_bf16_kernel<<<(int)(SQ / 4 + 255) / 256, 256, 0, stream>>>(wv, wv_b, (int)(SQ / 4));
  cvt_bf16_kernel<<<(int)(SQ / 4 + 255) / 256, 256, 0, stream>>>(wo, wo_b, (int)(SQ / 4));
  cvt_bf16_kernel<<<(int)(SF / 4 + 255) / 256, 256, 0, stream>>>(w1, w1_b, (int)(SF / 4));
  cvt_bf16_kernel<<<(int)(SF / 4 + 255) / 256, 256, 0, stream>>>(w3, w3_b, (int)(SF / 4));
  cvt_bf16_kernel<<<(int)(SF / 4 + 255) / 256, 256, 0, stream>>>(w2, w2_b, (int)(SF / 4));

  rmsnorm_kernel<<<NTOK, 256, 0, stream>>>(x, n1, h_b);

  dim3 gqkv(D_MODEL / 64, NTOK / 256);
  gemm_qkv_kernel<<<gqkv, 256, 0, stream>>>(h_b, wq_b, tpos, q_b, 0);
  gemm_qkv_kernel<<<gqkv, 256, 0, stream>>>(h_b, wk_b, tpos, k_b, 1);
  gemm_qkv_kernel<<<gqkv, 256, 0, stream>>>(h_b, wv_b, tpos, vt_b, 2);

  attn_kernel<<<512, 256, 0, stream>>>(q_b, k_b, vt_b, at_b);

  gemm_rc_f32_kernel<<<dim3(D_MODEL / 64, NTOK / 256), 256, 0, stream>>>(
      at_b, wo_b, x, y, D_MODEL, D_MODEL);

  rmsnorm_kernel<<<NTOK, 256, 0, stream>>>(y, n2, h2_b);

  gemm_ffn_gate_kernel<<<dim3(D_FF / 64, NTOK / 128), 256, 0, stream>>>(
      h2_b, w1_b, w3_b, f_b);

  gemm_rc_f32_kernel<<<dim3(D_MODEL / 64, NTOK / 256), 256, 0, stream>>>(
      f_b, w2_b, y, (float*)d_out, D_MODEL, D_FF);
}
